// HierarchicalMoE_50551765074004
// MI455X (gfx1250) — compile-verified, matrix operand order checked by reading, not run on hardware
//
#include <hip/hip_runtime.h>
#include <hip/hip_bf16.h>
#include <math.h>

// Problem dims (match reference)
constexpr int NB = 8, NS = 2048, NH = 512, ND = 1024, NE = 8;
constexpr int NT = NB * NS;          // 16384 tokens
constexpr int TILE_M = 64;           // tokens per workgroup
constexpr int DKCH  = 128;           // D chunk per iteration

typedef __attribute__((ext_vector_type(16))) __bf16 v16bf;
typedef __attribute__((ext_vector_type(8)))  float  v8f;
typedef __attribute__((ext_vector_type(4)))  __bf16 bf16x4;

// ---------------------------------------------------------------------------
// 1) elementwise f32 -> bf16 (x)
// ---------------------------------------------------------------------------
__global__ void cvt_bf16_x4(const float* __restrict__ src,
                            __bf16* __restrict__ dst, int n4) {
  int i = blockIdx.x * blockDim.x + threadIdx.x;
  if (i < n4) {
    float4 v = ((const float4*)src)[i];
    bf16x4 o;
    o.x = (__bf16)v.x; o.y = (__bf16)v.y; o.z = (__bf16)v.z; o.w = (__bf16)v.w;
    ((bf16x4*)dst)[i] = o;
  }
}

// ---------------------------------------------------------------------------
// 2) per-expert transpose + convert: src[e][R][C] f32  ->  dst[e][C][R] bf16
// ---------------------------------------------------------------------------
__global__ void transpose_cvt(const float* __restrict__ src,
                              __bf16* __restrict__ dst, int R, int C) {
  __shared__ float tile[32][33];
  const size_t eoff = (size_t)blockIdx.z * R * C;
  const float* s = src + eoff;
  __bf16*      d = dst + eoff;
  int tx = threadIdx.x, ty = threadIdx.y;
  int x = blockIdx.x * 32 + tx;
#pragma unroll
  for (int i = 0; i < 4; ++i) {
    int y = blockIdx.y * 32 + ty + 8 * i;
    tile[ty + 8 * i][tx] = s[(size_t)y * C + x];
  }
  __syncthreads();
  int r = blockIdx.y * 32 + tx;
#pragma unroll
  for (int i = 0; i < 4; ++i) {
    int c = blockIdx.x * 32 + ty + 8 * i;
    d[(size_t)c * R + r] = (__bf16)tile[tx][ty + 8 * i];
  }
}

// ---------------------------------------------------------------------------
// 3) router: probs[n][e] = softmax(x@Wt+bt)[e] * softmax(x@Wb+bb)[e]
//    one wave32 per token
// ---------------------------------------------------------------------------
__global__ void router_kernel(const float* __restrict__ x,
                              const float* __restrict__ Wt, const float* __restrict__ bt,
                              const float* __restrict__ Wb, const float* __restrict__ bb,
                              float* __restrict__ probs) {
  const int w = threadIdx.x >> 5, lane = threadIdx.x & 31;
  const int n = blockIdx.x * 8 + w;
  float at[8], ab[8];
#pragma unroll
  for (int e = 0; e < 8; ++e) { at[e] = 0.f; ab[e] = 0.f; }
  for (int h = lane; h < NH; h += 32) {
    float xv = x[(size_t)n * NH + h];
    float4 t0 = ((const float4*)&Wt[h * 8])[0];
    float4 t1 = ((const float4*)&Wt[h * 8])[1];
    float4 b0 = ((const float4*)&Wb[h * 8])[0];
    float4 b1v = ((const float4*)&Wb[h * 8])[1];
    at[0] += xv * t0.x; at[1] += xv * t0.y; at[2] += xv * t0.z; at[3] += xv * t0.w;
    at[4] += xv * t1.x; at[5] += xv * t1.y; at[6] += xv * t1.z; at[7] += xv * t1.w;
    ab[0] += xv * b0.x; ab[1] += xv * b0.y; ab[2] += xv * b0.z; ab[3] += xv * b0.w;
    ab[4] += xv * b1v.x; ab[5] += xv * b1v.y; ab[6] += xv * b1v.z; ab[7] += xv * b1v.w;
  }
#pragma unroll
  for (int off = 16; off > 0; off >>= 1) {
#pragma unroll
    for (int e = 0; e < 8; ++e) {
      at[e] += __shfl_xor(at[e], off);
      ab[e] += __shfl_xor(ab[e], off);
    }
  }
  float mt = -1e30f, mb = -1e30f;
#pragma unroll
  for (int e = 0; e < 8; ++e) {
    at[e] += bt[e]; ab[e] += bb[e];
    mt = fmaxf(mt, at[e]); mb = fmaxf(mb, ab[e]);
  }
  float st = 0.f, sb = 0.f;
#pragma unroll
  for (int e = 0; e < 8; ++e) {
    at[e] = expf(at[e] - mt); st += at[e];
    ab[e] = expf(ab[e] - mb); sb += ab[e];
  }
  if (lane == 0) {
    float inv = 1.f / (st * sb);
#pragma unroll
    for (int e = 0; e < 8; ++e) probs[(size_t)n * 8 + e] = at[e] * ab[e] * inv;
  }
}

// ---------------------------------------------------------------------------
// 4) fused MoE kernel: out[n,h] = sum_e [ (p_ne * gelu(x@W1_e + b1_e)) @ W2_e ][n,h]
//                                + sum_e p_ne * b2_e[h]
//    256 threads = 8 waves; block owns 64 tokens x full H output.
// ---------------------------------------------------------------------------
__global__ __launch_bounds__(256, 1) void moe_main(
    const __bf16* __restrict__ xb,    // [NT][NH] bf16
    const __bf16* __restrict__ w1t,   // [NE][ND][NH] bf16 (transposed)
    const __bf16* __restrict__ w2t,   // [NE][NH][ND] bf16 (transposed)
    const float*  __restrict__ probs, // [NT][NE]
    const float*  __restrict__ b1,    // [NE][ND]
    const float*  __restrict__ b2,    // [NE][NH]
    float*        __restrict__ out)   // [NT][NH]
{
  __shared__ __align__(32) __bf16 hidS[TILE_M * DKCH];   // 16 KB
  __shared__ __align__(16) float  probsS[TILE_M * NE];   // 2 KB

  const int tid  = threadIdx.x;
  const int w    = tid >> 5;
  const int lane = tid & 31;
  const int l16  = lane & 15;
  const int lhalf = lane >> 4;              // 0: lanes 0-15, 1: lanes 16-31
  const int tok0 = blockIdx.x * TILE_M;
  const int wm = w >> 2;                    // 0..1 : 32-row half of out tile
  const int wn = w & 3;                     // 0..3 : 128-col quarter of out tile

  for (int i = tid; i < TILE_M * NE; i += 256) probsS[i] = probs[(size_t)tok0 * NE + i];
  __syncthreads();

  v8f accO[2][8] = {};                      // out accumulators: 32x128 per wave

  for (int e = 0; e < NE; ++e) {
    for (int dc = 0; dc < ND / DKCH; ++dc) {
      // ---------------- GEMM1: hid[64 x 128] = x_tile @ W1t-chunk -------------
      // wave w owns d-columns [dq, dq+16), all 4 M-tiles
      const int dq = dc * DKCH + (w << 4);
      v8f accH[4] = {};
      const __bf16* bp = w1t + ((size_t)(e * ND + dq + l16)) * NH + (lhalf << 4);
      const __bf16* ap = xb + ((size_t)(tok0 + l16)) * NH + (lhalf << 4);
      for (int k = 0; k < NH; k += 32) {
        v16bf bf = *(const v16bf*)(bp + k);
        __builtin_prefetch(bp + k + 256, 0, 1);
#pragma unroll
        for (int mt = 0; mt < 4; ++mt) {
          v16bf af = *(const v16bf*)(ap + (size_t)mt * 16 * NH + k);
          accH[mt] = __builtin_amdgcn_wmma_f32_16x16x32_bf16(
              false, af, false, bf, (short)0, accH[mt], false, false);
        }
      }
      // epilogue: +b1, exact gelu, * routing prob, -> bf16 LDS
      const float b1v = b1[e * ND + dq + l16];
#pragma unroll
      for (int mt = 0; mt < 4; ++mt) {
#pragma unroll
        for (int j = 0; j < 8; ++j) {
          int rl = mt * 16 + j + (lhalf << 3);          // local token row
          float v = accH[mt][j] + b1v;
          float g = 0.5f * v * (1.0f + erff(v * 0.70710678118654752f));
          float p = probsS[rl * NE + e];
          hidS[rl * DKCH + (w << 4) + l16] = (__bf16)(g * p);
        }
      }
      __syncthreads();

      // ---------------- GEMM2: out[32 x 128 per wave] += hid @ W2t-chunk ------
      // preload all A-frags this wave needs (rows 32*wm..+32, K = 0..128)
      v16bf afr[2][4];
#pragma unroll
      for (int mt = 0; mt < 2; ++mt) {
        int rl = (wm << 5) + (mt << 4) + l16;
#pragma unroll
        for (int kk = 0; kk < 4; ++kk)
          afr[mt][kk] = *(const v16bf*)(&hidS[rl * DKCH + (kk << 5) + (lhalf << 4)]);
      }
      __syncthreads();   // all LDS reads done -> next iter may overwrite hidS

#pragma unroll
      for (int nt = 0; nt < 8; ++nt) {
        const int hc = (wn << 7) + (nt << 4) + l16;     // output column
        const __bf16* wp = w2t + ((size_t)(e * NH + hc)) * ND + dc * DKCH + (lhalf << 4);
        v16bf bf0 = *(const v16bf*)(wp);
        v16bf bf1 = *(const v16bf*)(wp + 32);
        v16bf bf2 = *(const v16bf*)(wp + 64);
        v16bf bf3 = *(const v16bf*)(wp + 96);
#pragma unroll
        for (int mt = 0; mt < 2; ++mt) {
          accO[mt][nt] = __builtin_amdgcn_wmma_f32_16x16x32_bf16(
              false, afr[mt][0], false, bf0, (short)0, accO[mt][nt], false, false);
          accO[mt][nt] = __builtin_amdgcn_wmma_f32_16x16x32_bf16(
              false, afr[mt][1], false, bf1, (short)0, accO[mt][nt], false, false);
          accO[mt][nt] = __builtin_amdgcn_wmma_f32_16x16x32_bf16(
              false, afr[mt][2], false, bf2, (short)0, accO[mt][nt], false, false);
          accO[mt][nt] = __builtin_amdgcn_wmma_f32_16x16x32_bf16(
              false, afr[mt][3], false, bf3, (short)0, accO[mt][nt], false, false);
        }
      }
    }
  }

  // final epilogue: add sum_e p[n,e]*b2[e,h], store f32
#pragma unroll
  for (int nt = 0; nt < 8; ++nt) {
    const int col = (wn << 7) + (nt << 4) + l16;
    float b2v[8];
#pragma unroll
    for (int e = 0; e < 8; ++e) b2v[e] = b2[e * NH + col];
#pragma unroll
    for (int mt = 0; mt < 2; ++mt) {
#pragma unroll
      for (int j = 0; j < 8; ++j) {
        int rl = (wm << 5) + (mt << 4) + j + (lhalf << 3);
        const float4* pv = (const float4*)&probsS[rl * NE];
        float4 p0 = pv[0], p1 = pv[1];
        float bias = p0.x * b2v[0] + p0.y * b2v[1] + p0.z * b2v[2] + p0.w * b2v[3] +
                     p1.x * b2v[4] + p1.y * b2v[5] + p1.z * b2v[6] + p1.w * b2v[7];
        out[(size_t)(tok0 + rl) * NH + col] = accO[mt][nt][j] + bias;
      }
    }
  }
}

// ---------------------------------------------------------------------------
extern "C" void kernel_launch(void* const* d_in, const int* in_sizes, int n_in,
                              void* d_out, int out_size, void* d_ws, size_t ws_size,
                              hipStream_t stream) {
  const float* x  = (const float*)d_in[0];  // [NB,NS,NH]
  const float* W1 = (const float*)d_in[1];  // [NE,NH,ND]
  const float* b1 = (const float*)d_in[2];  // [NE,ND]
  const float* W2 = (const float*)d_in[3];  // [NE,ND,NH]
  const float* b2 = (const float*)d_in[4];  // [NE,NH]
  const float* Wt = (const float*)d_in[5];  // [NH,NE]
  const float* bt = (const float*)d_in[6];  // [NE]
  const float* Wb = (const float*)d_in[7];  // [NH,NE]
  const float* bb = (const float*)d_in[8];  // [NE]
  float* out = (float*)d_out;

  // workspace layout
  __bf16* xb   = (__bf16*)d_ws;                      // NT*NH
  __bf16* w1t  = xb  + (size_t)NT * NH;              // NE*ND*NH
  __bf16* w2t  = w1t + (size_t)NE * ND * NH;         // NE*NH*ND
  float*  prbs = (float*)(w2t + (size_t)NE * NH * ND); // NT*NE

  // 1) x -> bf16
  {
    int n4 = NT * NH / 4;
    cvt_bf16_x4<<<(n4 + 255) / 256, 256, 0, stream>>>(x, xb, n4);
  }
  // 2) W1[e]: [NH][ND] -> W1t[e]: [ND][NH] bf16
  transpose_cvt<<<dim3(ND / 32, NH / 32, NE), dim3(32, 8), 0, stream>>>(W1, w1t, NH, ND);
  // 3) W2[e]: [ND][NH] -> W2t[e]: [NH][ND] bf16
  transpose_cvt<<<dim3(NH / 32, ND / 32, NE), dim3(32, 8), 0, stream>>>(W2, w2t, ND, NH);
  // 4) routing probabilities
  router_kernel<<<NT / 8, 256, 0, stream>>>(x, Wt, bt, Wb, bb, prbs);
  // 5) fused dense-MoE
  moe_main<<<NT / TILE_M, 256, 0, stream>>>(xb, w1t, w2t, prbs, b1, b2, out);
}
